// PhaserModel_17300128269093
// MI455X (gfx1250) — matrix operand order repaired and checked by reading, not compiled, not run on hardware
//
#include <hip/hip_runtime.h>
#include <hip/hip_bf16.h>
#include <math.h>

#define SR_F 44100.0f
#define KSTAGES 4
#define HW 32
#define TSAMP 262144
#define TILES_PER_CH (TSAMP/16)      // 16384 = 2^14
#define TOTAL_TILES (2*TILES_PER_CH) // 32768

typedef __attribute__((ext_vector_type(16))) _Float16 v16h;
typedef __attribute__((ext_vector_type(8)))  float    v8f;

union V16H { v16h v; unsigned int u[8]; };

// Branch-free fast tanh: tanh(x) = 1 - 2/(exp(2x)+1).
// v_exp_f32 + v_rcp_f32; saturates correctly for |x| large.
__device__ __forceinline__ float tanh_fast(float x) {
    const float e = __expf(2.0f * x);
    return 1.0f - 2.0f * __builtin_amdgcn_rcpf(e + 1.0f);
}

// ---------------------------------------------------------------------------
// Phase 1: per-sample coefficient network.
//   z = amp*cos(arg);  h1 = tanh(z*W1+b1)  [VALU/TRANS, branch-free]
//   h2 = tanh(h1@W2 + b2)                  [2x v_wmma_f32_16x16x32_f16]
//   m  = h2@W3 + b3                        [shfl reduction]
//   p  = tanh((1-tan(d))/(1+tan(d))), d = bias + depth*0.5*(1+m)
// One wave handles a 16-sample tile; grid-stride over 32768 tiles.
// ---------------------------------------------------------------------------
__global__ __launch_bounds__(256)
void phaser_coeff_wmma(const float* __restrict__ lfo_rate,
                       const float* __restrict__ phoff,
                       const float* __restrict__ amp,
                       const float* __restrict__ bias,
                       const float* __restrict__ depth,
                       const float* __restrict__ W1,
                       const float* __restrict__ b1,
                       const float* __restrict__ W2,
                       const float* __restrict__ b2,
                       const float* __restrict__ W3,
                       const float* __restrict__ b3,
                       float* __restrict__ pOut)
{
    // W2 transposed (w2t[out][in]) so B-fragment K-pairs are contiguous u32s.
    __shared__ _Float16 w2t[32][36];        // padded stride vs bank conflicts
    __shared__ _Float16 h1s[8][16][36];     // per-wave layer-1 staging

    const int tid = threadIdx.x;
    for (int e = tid; e < 1024; e += 256) {
        const int i = e >> 5, j = e & 31;   // W2[i][j]: i = in-feature, j = out
        w2t[j][i] = (_Float16)W2[e];
    }
    __syncthreads();

    const int lane = tid & 31;
    const int wv   = tid >> 5;
    const int n0   = lane & 15;
    const int hi   = lane >> 4;

    // Pack B fragments once per wave. Lane holds column n, K-pairs in u32s.
    // lanes 0-15: K=0..15 ; lanes 16-31: K=16..31 (contiguous order, matching A).
    V16H B0, B1;
    {
        const int kb = hi * 16;
        for (int j = 0; j < 8; ++j) {
            B0.u[j] = *(const unsigned int*)&w2t[n0][kb + 2*j];
            B1.u[j] = *(const unsigned int*)&w2t[16 + n0][kb + 2*j];
        }
    }

    const float stepv  = 2.0f * 3.14159265358979323846f * lfo_rate[0] / SR_F;
    const float po     = phoff[0];
    const float ampv   = amp[0], biasv = bias[0], depthv = depth[0];
    const float b3v    = b3[0];
    const int   j0     = hi * 16;            // this lane's 16 hidden units
    float w1r[16], b1r[16];
    for (int i = 0; i < 16; ++i) { w1r[i] = W1[j0 + i]; b1r[i] = b1[j0 + i]; }
    const float w3a = W3[n0],  w3b = W3[16 + n0];
    const float b2a = b2[n0],  b2b = b2[16 + n0];

    const int gwave  = blockIdx.x * (blockDim.x >> 5) + wv;
    const int nwaves = gridDim.x * (blockDim.x >> 5);

    for (int tt = gwave; tt < TOTAL_TILES; tt += nwaves) {
        const int ch  = tt >> 14;                        // channel
        const int t0  = (tt & (TILES_PER_CH - 1)) << 4;  // first sample of tile
        const int m   = n0;                              // sample row this lane owns
        const int idx = t0 + m;

        const float z = ampv * __cosf(stepv * (float)(idx + 1) + (ch ? po : 0.0f));

        // layer 1: 16 hidden units for sample m, packed f16 pairs to LDS
        for (int i = 0; i < 8; ++i) {
            const _Float16 ha = (_Float16)tanh_fast(z * w1r[2*i]     + b1r[2*i]);
            const _Float16 hb = (_Float16)tanh_fast(z * w1r[2*i + 1] + b1r[2*i + 1]);
            const unsigned int pk =
                (unsigned int)__builtin_bit_cast(unsigned short, ha) |
                ((unsigned int)__builtin_bit_cast(unsigned short, hb) << 16);
            *(unsigned int*)&h1s[wv][m][j0 + 2*i] = pk;
        }
        asm volatile("s_wait_dscnt 0" ::: "memory"); // cross-lane LDS ordering

        // Gather A fragment: lane(L) row=L&15; lanes 0-15: K=0-15,
        // lanes 16-31: K=16-31 (contiguous order, matching B).
        V16H A;
        const int kb = hi * 16;
        for (int i = 0; i < 4; ++i) {
            A.u[i]     = *(const unsigned int*)&h1s[wv][m][kb + 2*i];
            A.u[4 + i] = *(const unsigned int*)&h1s[wv][m][kb + 8 + 2*i];
        }

        v8f cz = {};
        v8f c0 = __builtin_amdgcn_wmma_f32_16x16x32_f16(
            false, A.v, false, B0.v, (short)0, cz, false, false);
        v8f c1 = __builtin_amdgcn_wmma_f32_16x16x32_f16(
            false, A.v, false, B1.v, (short)0, cz, false, false);

        // layer-2 activation + layer-3 partial products.
        // C fragment: c[v] = row (v + 8*hi), col n0 (c0) / 16+n0 (c1).
        float tsum[8];
        for (int v = 0; v < 8; ++v)
            tsum[v] = tanh_fast(c0[v] + b2a) * w3a + tanh_fast(c1[v] + b2b) * w3b;

        // reduce over the 16 columns held within each 16-lane half
        for (int msk = 1; msk < 16; msk <<= 1)
            for (int v = 0; v < 8; ++v)
                tsum[v] += __shfl_xor(tsum[v], msk, 32);

        // Spread the tail transcendentals: after the reduction every lane in a
        // 16-lane half holds all 8 row sums; lanes n0<8 take one sample each.
        if (n0 < 8) {
            const float mval = tsum[n0] + b3v;
            const float d    = biasv + depthv * 0.5f * (1.0f + mval);
            const float td   = tanf(d);
            const float pc   = tanh_fast((1.0f - td) * __builtin_amdgcn_rcpf(1.0f + td));
            pOut[ch * TSAMP + t0 + hi * 8 + n0] = pc;
        }
    }
}

// ---------------------------------------------------------------------------
// Phase 2: K=4 cascaded time-varying all-pass stages as chunked affine scans.
// Stage recurrence: y[n] = (-p[n])*y[n-1] + (p[n]*s[n] + s[n-1]).
// One 1024-thread workgroup per channel; 256 samples/thread.
// ---------------------------------------------------------------------------
__global__ __launch_bounds__(1024)
void phaser_scan(const float* __restrict__ x,
                 const float* __restrict__ p,
                 float* __restrict__ buf0,
                 float* __restrict__ buf1,
                 float* __restrict__ out)
{
    __shared__ float As[1024];
    __shared__ float Ys[1024];

    const int ch    = blockIdx.x;
    const int t     = threadIdx.x;
    const int CH    = TSAMP / 1024;       // 256 samples per thread
    const int start = t * CH;
    const float* pc = p + ch * TSAMP;

    for (int stage = 0; stage < KSTAGES; ++stage) {
        const float* sIn =
            (stage == 0) ? x :
            (stage == 1) ? buf0 + ch * TSAMP :
            (stage == 2) ? buf1 + ch * TSAMP : buf0 + ch * TSAMP;
        float* yOut =
            (stage == 0) ? buf0 + ch * TSAMP :
            (stage == 1) ? buf1 + ch * TSAMP :
            (stage == 2) ? buf0 + ch * TSAMP : nullptr;

        // Pass A: reduce chunk to affine map (Acc, Ycc) with y_in = 0.
        float Acc = 1.0f, Ycc = 0.0f;
        float sp = (start == 0) ? 0.0f : sIn[start - 1];
        for (int i = 0; i < CH; ++i) {
            const int   n   = start + i;
            const float pv  = pc[n];
            const float cur = sIn[n];
            Ycc = (pv * cur + sp) - pv * Ycc;
            Acc *= -pv;
            sp = cur;
        }
        As[t] = Acc; Ys[t] = Ycc;
        __syncthreads();

        // Inclusive Hillis-Steele scan of affine maps (compose prev-then-cur).
        for (int off = 1; off < 1024; off <<= 1) {
            float aP = 0.0f, yP = 0.0f;
            const bool has = (t >= off);
            if (has) { aP = As[t - off]; yP = Ys[t - off]; }
            __syncthreads();
            if (has) { Ys[t] = As[t] * yP + Ys[t]; As[t] = As[t] * aP; }
            __syncthreads();
        }
        const float carry = (t == 0) ? 0.0f : Ys[t - 1];
        __syncthreads();

        // Pass C: replay chunk with correct carry.
        float y = carry;
        sp = (start == 0) ? 0.0f : sIn[start - 1];
        if (stage < KSTAGES - 1) {
            for (int i = 0; i < CH; ++i) {
                const int   n   = start + i;
                const float pv  = pc[n];
                const float cur = sIn[n];
                y = pv * cur + sp - pv * y;
                yOut[n] = y;
                sp = cur;
            }
        } else {
            for (int i = 0; i < CH; ++i) {
                const int   n   = start + i;
                const float pv  = pc[n];
                const float cur = sIn[n];
                y = pv * cur + sp - pv * y;
                out[ch * TSAMP + n] = 0.5f * (x[n] + y);  // fused dry/wet mix
                sp = cur;
            }
        }
        __threadfence_block();   // make global stage output visible in-block
        __syncthreads();
    }
}

// ---------------------------------------------------------------------------
extern "C" void kernel_launch(void* const* d_in, const int* in_sizes, int n_in,
                              void* d_out, int out_size, void* d_ws, size_t ws_size,
                              hipStream_t stream) {
    const float* x        = (const float*)d_in[0];
    const float* lfo_rate = (const float*)d_in[1];
    const float* phoff    = (const float*)d_in[2];
    const float* amp      = (const float*)d_in[3];
    const float* bias     = (const float*)d_in[4];
    const float* depth    = (const float*)d_in[5];
    const float* W1       = (const float*)d_in[6];
    const float* b1       = (const float*)d_in[7];
    const float* W2       = (const float*)d_in[8];
    const float* b2       = (const float*)d_in[9];
    const float* W3       = (const float*)d_in[10];
    const float* b3       = (const float*)d_in[11];

    float* p    = (float*)d_ws;          // [2*T] all-pass coefficients
    float* buf0 = p    + 2 * TSAMP;      // [2*T] stage ping
    float* buf1 = buf0 + 2 * TSAMP;      // [2*T] stage pong

    phaser_coeff_wmma<<<512, 256, 0, stream>>>(
        lfo_rate, phoff, amp, bias, depth, W1, b1, W2, b2, W3, b3, p);

    phaser_scan<<<2, 1024, 0, stream>>>(x, p, buf0, buf1, (float*)d_out);
}
